// STRec_Model_27358941676177
// MI455X (gfx1250) — matrix operand order repaired and checked by reading, not compiled, not run on hardware
//
#include <hip/hip_runtime.h>

// ---------------------------------------------------------------------------
// MoE reprogramming attention for MI455X (gfx1250, wave32, WMMA).
//
// Roofline: ~106 GFLOP vs ~70 MB unique traffic (~3us @ 23.3 TB/s) -> compute
// bound by >30x, so everything runs through v_wmma_f32_16x16x32_bf16.
// All f32->bf16 conversion + weight transposition is hoisted into one-off
// bandwidth-bound prep kernels; the hot GEMM/attention loops are pure
// (2x global_load_b128 per fragment) + WMMA. Experts processed sequentially;
// workspace ~42.6 MB, L2-resident (192 MB).
// ---------------------------------------------------------------------------

#define B_   16
#define L_   256
#define D_   512
#define H_   8
#define DK_  64
#define S_   2048
#define HD_  512
#define E_   4
#define ROWS (B_ * L_)   // 4096

typedef __attribute__((ext_vector_type(16))) __bf16 v16bf;
typedef __attribute__((ext_vector_type(8)))  float  v8f;

__device__ __forceinline__ unsigned short f2bf(float f) {
  union { __bf16 h; unsigned short u; } c;
  c.h = (__bf16)f;              // native cvt on gfx1250
  return c.u;
}

__device__ __forceinline__ v8f wmma_bf16(v16bf a, v16bf b, v8f c) {
  // (neg_a, A, neg_b, B, c_mod, C, reuse_a, reuse_b)
  return __builtin_amdgcn_wmma_f32_16x16x32_bf16(false, a, false, b, (short)0, c,
                                                 false, false);
}

// ---- Fragment loaders (wave32, contiguous k order shared by A and B) ---------------
// A fragment 16x32 (MxK), row-major bf16 source: elem (m,k) = src[m*ld + k].
// Lane layout: lanes 0-15 row M=lane, K 0..15; lanes 16-31 same
// row, K 16..31. k-pairs are contiguous -> 2x 16-byte loads per lane.
__device__ __forceinline__ v16bf load_a_bf16(const unsigned short* src, int ld) {
  const int lane = threadIdx.x & 31;
  const int m = lane & 15;
  const int kh = (lane >> 4) << 4;              // 0 or 16
  const unsigned short* p = src + (size_t)m * ld + kh;
  union { v16bf v; unsigned int w[8]; } f;
#pragma unroll
  for (int i = 0; i < 4; ++i) f.w[i]     = *(const unsigned int*)(p + 2 * i);
#pragma unroll
  for (int i = 0; i < 4; ++i) f.w[4 + i] = *(const unsigned int*)(p + 8 + 2 * i);
  return f.v;
}

// B fragment 32x16 (KxN) from K-contiguous ("transposed") source:
// elem (k,n) = src[n*ld + k]. Lane n: col N=n, K=kb..kb+15 contiguous
// -> 2x 16-byte loads per lane.
__device__ __forceinline__ v16bf load_b_bf16_cm(const unsigned short* src, int ld) {
  const int lane = threadIdx.x & 31;
  const int n = lane & 15;
  const int kb = (lane >> 4) << 4;              // 0 or 16
  const uint4* p = (const uint4*)(src + (size_t)n * ld + kb);
  union { v16bf v; uint4 q[2]; } f;
  f.q[0] = p[0];
  f.q[1] = p[1];
  return f.v;
}

// ---------------------------------------------------------------------------
// Prep kernel A: bulk f32 -> bf16 (4 elems/thread, bandwidth bound)
// ---------------------------------------------------------------------------
__global__ void cvt_bf16(const float* __restrict__ in, unsigned short* __restrict__ out,
                         int n4) {
  const int i = blockIdx.x * blockDim.x + threadIdx.x;
  if (i >= n4) return;
  const float4 v = *(const float4*)(in + (size_t)i * 4);
  ushort4 o;
  o.x = f2bf(v.x); o.y = f2bf(v.y); o.z = f2bf(v.z); o.w = f2bf(v.w);
  *(ushort4*)(out + (size_t)i * 4) = o;
}

// ---------------------------------------------------------------------------
// Prep kernel B: W_f32[K,N] -> Wt_bf16[N,K] (16x16 LDS-tiled transpose)
// ---------------------------------------------------------------------------
__global__ void cvt_transpose_bf16(const float* __restrict__ in,
                                   unsigned short* __restrict__ out, int K, int N) {
  __shared__ float tile[16][17];
  const int tx = threadIdx.x & 15, ty = threadIdx.x >> 4;
  const int k0 = blockIdx.y * 16, n0 = blockIdx.x * 16;
  tile[ty][tx] = in[(size_t)(k0 + ty) * N + n0 + tx];
  __syncthreads();
  out[(size_t)(n0 + ty) * K + k0 + tx] = f2bf(tile[tx][ty]);
}

// ---------------------------------------------------------------------------
// Kernel 1: router gates = softmax(x @ gate_W + gate_b), E=4. Tiny.
// ---------------------------------------------------------------------------
__global__ void gate_softmax(const float* __restrict__ x, const float* __restrict__ gW,
                             const float* __restrict__ gb, float* __restrict__ gates) {
  const int row = blockIdx.x * blockDim.x + threadIdx.x;
  if (row >= ROWS) return;
  const float* xr = x + (size_t)row * D_;
  float a0 = gb[0], a1 = gb[1], a2 = gb[2], a3 = gb[3];
  for (int k = 0; k < D_; k += 4) {
    const float4 xv = *(const float4*)(xr + k);
    const float4 w0 = *(const float4*)(gW + (size_t)k * 4);
    const float4 w1 = *(const float4*)(gW + (size_t)(k + 1) * 4);
    const float4 w2 = *(const float4*)(gW + (size_t)(k + 2) * 4);
    const float4 w3 = *(const float4*)(gW + (size_t)(k + 3) * 4);
    a0 += xv.x * w0.x + xv.y * w1.x + xv.z * w2.x + xv.w * w3.x;
    a1 += xv.x * w0.y + xv.y * w1.y + xv.z * w2.y + xv.w * w3.y;
    a2 += xv.x * w0.z + xv.y * w1.z + xv.z * w2.z + xv.w * w3.z;
    a3 += xv.x * w0.w + xv.y * w1.w + xv.z * w2.w + xv.w * w3.w;
  }
  const float m = fmaxf(fmaxf(a0, a1), fmaxf(a2, a3));
  const float e0 = __expf(a0 - m), e1 = __expf(a1 - m);
  const float e2 = __expf(a2 - m), e3 = __expf(a3 - m);
  const float inv = 1.0f / (e0 + e1 + e2 + e3);
  *(float4*)(gates + (size_t)row * 4) = make_float4(e0 * inv, e1 * inv, e2 * inv, e3 * inv);
}

// ---------------------------------------------------------------------------
// Kernel 2: C[M,N] = A_bf16[M,K] @ Bt_bf16[N,K]^T + bias.
// One wave computes a 16x64 slab: A fragment reused across 4 WMMAs.
// transpose_out=1 stores C^T (used for V so attention B-frags are contiguous).
// ---------------------------------------------------------------------------
__device__ __forceinline__ void store_tile(v8f c, unsigned short* __restrict__ C,
                                           const float* __restrict__ bias,
                                           int M, int N, int tm, int ncol0,
                                           int transpose_out) {
  const int lane = threadIdx.x & 31;
  const int n = lane & 15;
  const int mb = (lane >> 4) << 3;
  const int ncol = ncol0 + n;
  const float bv = bias[ncol];
#pragma unroll
  for (int j = 0; j < 8; ++j) {
    const unsigned short bf = f2bf(c[j] + bv);
    const int row = tm * 16 + mb + j;
    if (transpose_out) C[(size_t)ncol * M + row] = bf;
    else               C[(size_t)row * N + ncol] = bf;
  }
}

__global__ void __launch_bounds__(128) gemm_bf16(
    const unsigned short* __restrict__ A, const unsigned short* __restrict__ Bt,
    const float* __restrict__ bias, unsigned short* __restrict__ C,
    int M, int N, int K, int transpose_out) {
  const int wave = threadIdx.x >> 5;
  const int n64 = blockIdx.x * 4 + wave;        // 64-column slab index
  const int tm = blockIdx.y;                    // 16-row tile index
  const unsigned short* Ab = A + (size_t)tm * 16 * K;
  const unsigned short* Bb = Bt + (size_t)n64 * 64 * K;
  v8f c0 = {}, c1 = {}, c2 = {}, c3 = {};
  for (int k0 = 0; k0 < K; k0 += 32) {
    const v16bf a = load_a_bf16(Ab + k0, K);
    c0 = wmma_bf16(a, load_b_bf16_cm(Bb + k0, K), c0);
    c1 = wmma_bf16(a, load_b_bf16_cm(Bb + (size_t)16 * K + k0, K), c1);
    c2 = wmma_bf16(a, load_b_bf16_cm(Bb + (size_t)32 * K + k0, K), c2);
    c3 = wmma_bf16(a, load_b_bf16_cm(Bb + (size_t)48 * K + k0, K), c3);
  }
  store_tile(c0, C, bias, M, N, tm, n64 * 64,      transpose_out);
  store_tile(c1, C, bias, M, N, tm, n64 * 64 + 16, transpose_out);
  store_tile(c2, C, bias, M, N, tm, n64 * 64 + 32, transpose_out);
  store_tile(c3, C, bias, M, N, tm, n64 * 64 + 48, transpose_out);
}

// ---------------------------------------------------------------------------
// Kernel 3: flash attention. One wave owns a 16-row L tile of one (b,h):
// streams S in chunks of 32: 4 score WMMAs + online softmax + 4 AV WMMAs.
// K staged row-major [S,HD] (contiguous along DK), V staged transposed
// [HD,S] (contiguous along S) -> every B fragment is 2x b128 per lane.
// P (exp'd scores) is re-shaped C-layout -> A-layout through per-wave LDS.
// ---------------------------------------------------------------------------
__global__ void __launch_bounds__(128) attn_flash(
    const unsigned short* __restrict__ Q,   // [B*L, HD] bf16
    const unsigned short* __restrict__ Km,  // [S, HD]   bf16
    const unsigned short* __restrict__ Vt,  // [HD, S]   bf16 (transposed)
    unsigned short* __restrict__ O) {       // [B*L, HD] bf16
  __shared__ __align__(16) unsigned short Pbuf[4][16 * 32];

  const int wave = threadIdx.x >> 5;
  const int lane = threadIdx.x & 31;
  const int bh = blockIdx.z;
  const int b = bh >> 3, h = bh & 7;
  const int row0 = (blockIdx.y * 4 + wave) * 16;

  const unsigned short* Qb = Q + (size_t)(b * L_ + row0) * HD_ + h * DK_;
  const unsigned short* Kb = Km + (size_t)h * DK_;
  const unsigned short* Vb = Vt + (size_t)(h * DK_) * S_;

  const v16bf qa0 = load_a_bf16(Qb, HD_);       // DK 0..31
  const v16bf qa1 = load_a_bf16(Qb + 32, HD_);  // DK 32..63

  float mrow[8], lrow[8];
  v8f o0 = {}, o1 = {}, o2 = {}, o3 = {};
#pragma unroll
  for (int j = 0; j < 8; ++j) { mrow[j] = -1e30f; lrow[j] = 0.0f; }

  const int n = lane & 15;
  const int mb = (lane >> 4) << 3;
  unsigned short* P = &Pbuf[wave][0];

  for (int s0 = 0; s0 < S_; s0 += 32) {
    v8f sc0 = {}, sc1 = {};
    sc0 = wmma_bf16(qa0, load_b_bf16_cm(Kb + (size_t)s0 * HD_,             HD_), sc0);
    sc0 = wmma_bf16(qa1, load_b_bf16_cm(Kb + (size_t)s0 * HD_ + 32,        HD_), sc0);
    sc1 = wmma_bf16(qa0, load_b_bf16_cm(Kb + (size_t)(s0 + 16) * HD_,      HD_), sc1);
    sc1 = wmma_bf16(qa1, load_b_bf16_cm(Kb + (size_t)(s0 + 16) * HD_ + 32, HD_), sc1);
#pragma unroll
    for (int j = 0; j < 8; ++j) { sc0[j] *= 0.125f; sc1[j] *= 0.125f; }

    // online softmax; row j lives across the 16 lanes of this half-wave
#pragma unroll
    for (int j = 0; j < 8; ++j) {
      float v = fmaxf(sc0[j], sc1[j]);
#pragma unroll
      for (int off = 1; off < 16; off <<= 1) v = fmaxf(v, __shfl_xor(v, off, 32));
      const float mn = fmaxf(mrow[j], v);
      const float corr = __expf(mrow[j] - mn);
      mrow[j] = mn;
      const float p0 = __expf(sc0[j] - mn);
      const float p1 = __expf(sc1[j] - mn);
      float ps = p0 + p1;
#pragma unroll
      for (int off = 1; off < 16; off <<= 1) ps += __shfl_xor(ps, off, 32);
      lrow[j] = lrow[j] * corr + ps;
      o0[j] *= corr; o1[j] *= corr; o2[j] *= corr; o3[j] *= corr;
      const int r = mb + j;
      P[r * 32 + n]      = f2bf(p0);
      P[r * 32 + 16 + n] = f2bf(p1);
    }
    // DS ops are in-order within a wave; make the cross-lane handoff explicit
    asm volatile("s_wait_dscnt 0" ::: "memory");

    const v16bf pa = load_a_bf16(P, 32);        // A-layout reload of P
    o0 = wmma_bf16(pa, load_b_bf16_cm(Vb + s0,                   S_), o0);
    o1 = wmma_bf16(pa, load_b_bf16_cm(Vb + (size_t)16 * S_ + s0, S_), o1);
    o2 = wmma_bf16(pa, load_b_bf16_cm(Vb + (size_t)32 * S_ + s0, S_), o2);
    o3 = wmma_bf16(pa, load_b_bf16_cm(Vb + (size_t)48 * S_ + s0, S_), o3);
    asm volatile("s_wait_dscnt 0" ::: "memory"); // P consumed before overwrite
  }

  unsigned short* Ob = O + (size_t)(b * L_ + row0) * HD_ + h * DK_;
#pragma unroll
  for (int j = 0; j < 8; ++j) {
    const float inv = 1.0f / lrow[j];
    unsigned short* orow = Ob + (size_t)(mb + j) * HD_;
    orow[n]      = f2bf(o0[j] * inv);
    orow[16 + n] = f2bf(o1[j] * inv);
    orow[32 + n] = f2bf(o2[j] * inv);
    orow[48 + n] = f2bf(o3[j] * inv);
  }
}

// ---------------------------------------------------------------------------
// Kernel 4: out (+)= gates[:,e] * (AO_bf16 @ WoT^T + bo). One wave per tile.
// ---------------------------------------------------------------------------
__global__ void __launch_bounds__(128) out_proj(
    const unsigned short* __restrict__ AO, const unsigned short* __restrict__ WoT,
    const float* __restrict__ bo, const float* __restrict__ gates,
    float* __restrict__ out, int e, int accumulate) {
  const int wave = threadIdx.x >> 5;        // N tile 0..3 (64 cols total)
  const int tm = blockIdx.x;                // M tile 0..255
  const unsigned short* Ab = AO + (size_t)tm * 16 * HD_;
  const unsigned short* Bb = WoT + (size_t)wave * 16 * HD_;
  v8f c = {};
  for (int k0 = 0; k0 < HD_; k0 += 32) {
    const v16bf a = load_a_bf16(Ab + k0, HD_);
    c = wmma_bf16(a, load_b_bf16_cm(Bb + k0, HD_), c);
  }
  const int lane = threadIdx.x & 31;
  const int n = lane & 15;
  const int mb = (lane >> 4) << 3;
  const float bv = bo[wave * 16 + n];
#pragma unroll
  for (int j = 0; j < 8; ++j) {
    const int row = tm * 16 + mb + j;
    const float g = gates[(size_t)row * 4 + e];
    const float val = g * (c[j] + bv);
    float* dst = out + (size_t)row * 64 + wave * 16 + n;
    if (accumulate) *dst += val; else *dst = val;
  }
}

// ---------------------------------------------------------------------------
extern "C" void kernel_launch(void* const* d_in, const int* in_sizes, int n_in,
                              void* d_out, int out_size, void* d_ws, size_t ws_size,
                              hipStream_t stream) {
  const float* seq  = (const float*)d_in[0];
  const float* gW   = (const float*)d_in[1];
  const float* gb   = (const float*)d_in[2];
  const float* src0 = (const float*)d_in[3];
  const float* q0W  = (const float*)d_in[4];
  const float* q0b  = (const float*)d_in[5];
  const float* k0W  = (const float*)d_in[6];
  const float* k0b  = (const float*)d_in[7];
  const float* v0W  = (const float*)d_in[8];
  const float* v0b  = (const float*)d_in[9];
  const float* o0W  = (const float*)d_in[10];
  const float* o0b  = (const float*)d_in[11];
  const float* srcR = (const float*)d_in[12];
  const float* qrW  = (const float*)d_in[13];
  const float* qrb  = (const float*)d_in[14];
  const float* krW  = (const float*)d_in[15];
  const float* krb  = (const float*)d_in[16];
  const float* vrW  = (const float*)d_in[17];
  const float* vrb  = (const float*)d_in[18];
  const float* orW  = (const float*)d_in[19];
  const float* orb  = (const float*)d_in[20];
  float* out = (float*)d_out;

  // workspace layout (bf16 staging; ~42.6 MB total, L2-resident)
  unsigned short* seqB = (unsigned short*)d_ws;                 // [4096,512]
  unsigned short* srcB = seqB + (size_t)ROWS * D_;              // [2048,4096] max
  unsigned short* WqT  = srcB + (size_t)S_ * 4096;              // [512,512]
  unsigned short* WkT  = WqT + (size_t)HD_ * D_;                // [512,4096] max
  unsigned short* WvT  = WkT + (size_t)HD_ * 4096;              // [512,4096] max
  unsigned short* WoT  = WvT + (size_t)HD_ * 4096;              // [64,512]
  unsigned short* Qws  = WoT + (size_t)64 * HD_;                // [4096,512]
  unsigned short* Kws  = Qws + (size_t)ROWS * HD_;              // [2048,512]
  unsigned short* VtWs = Kws + (size_t)S_ * HD_;                // [512,2048] (HD,S)
  unsigned short* AOws = VtWs + (size_t)HD_ * S_;               // [4096,512]
  float* gates = (float*)(AOws + (size_t)ROWS * HD_);           // [4096,4]

  gate_softmax<<<ROWS / 256, 256, 0, stream>>>(seq, gW, gb, gates);

  // seq_emb -> bf16 once
  cvt_bf16<<<(ROWS * D_ / 4) / 256, 256, 0, stream>>>(seq, seqB, ROWS * D_ / 4);

  for (int e = 0; e < E_; ++e) {
    const float* Wq  = e ? qrW + (size_t)(e - 1) * D_ * HD_  : q0W;
    const float* bq  = e ? qrb + (size_t)(e - 1) * HD_       : q0b;
    const int  dllm  = e ? 768 : 4096;
    const float* Wk  = e ? krW + (size_t)(e - 1) * 768 * HD_ : k0W;
    const float* bk  = e ? krb + (size_t)(e - 1) * HD_       : k0b;
    const float* Wv  = e ? vrW + (size_t)(e - 1) * 768 * HD_ : v0W;
    const float* bvp = e ? vrb + (size_t)(e - 1) * HD_       : v0b;
    const float* src = e ? srcR + (size_t)(e - 1) * S_ * 768 : src0;
    const float* Wo  = e ? orW + (size_t)(e - 1) * HD_ * 64  : o0W;
    const float* bo  = e ? orb + (size_t)(e - 1) * 64        : o0b;

    // prep: bf16 source + transposed bf16 weights for this expert
    cvt_bf16<<<(S_ * dllm / 4) / 256, 256, 0, stream>>>(src, srcB, S_ * dllm / 4);
    cvt_transpose_bf16<<<dim3(HD_ / 16, D_ / 16),  256, 0, stream>>>(Wq, WqT, D_,  HD_);
    cvt_transpose_bf16<<<dim3(HD_ / 16, dllm / 16), 256, 0, stream>>>(Wk, WkT, dllm, HD_);
    cvt_transpose_bf16<<<dim3(HD_ / 16, dllm / 16), 256, 0, stream>>>(Wv, WvT, dllm, HD_);
    cvt_transpose_bf16<<<dim3(64 / 16, HD_ / 16),  256, 0, stream>>>(Wo, WoT, HD_, 64);

    // projections (all WMMA, pure bf16 streams)
    gemm_bf16<<<dim3(HD_ / 256, ROWS / 16), 128, 0, stream>>>(seqB, WqT, bq, Qws,
                                                              ROWS, HD_, D_, 0);
    gemm_bf16<<<dim3(HD_ / 256, S_ / 16), 128, 0, stream>>>(srcB, WkT, bk, Kws,
                                                            S_, HD_, dllm, 0);
    gemm_bf16<<<dim3(HD_ / 256, S_ / 16), 128, 0, stream>>>(srcB, WvT, bvp, VtWs,
                                                            S_, HD_, dllm, 1);
    attn_flash<<<dim3(1, L_ / 64, B_ * H_), 128, 0, stream>>>(Qws, Kws, VtWs, AOws);
    out_proj<<<dim3(ROWS / 16), 128, 0, stream>>>(AOws, WoT, bo, gates, out, e, e > 0);
  }
}
